// LSTM_52338471469773
// MI455X (gfx1250) — matrix operand order repaired and checked by reading, not compiled, not run on hardware
//
#include <hip/hip_runtime.h>
#include <hip/hip_bf16.h>

typedef __bf16 bf16;
typedef __attribute__((ext_vector_type(16))) __bf16 v16bf;
typedef __attribute__((ext_vector_type(8)))  float  v8f;
typedef __attribute__((ext_vector_type(4)))  unsigned int u32x4;
typedef __attribute__((ext_vector_type(8)))  int    i32x8;
typedef __attribute__((ext_vector_type(4)))  int    i32x4;

#define B_   32
#define T_   1024
#define D_   512
#define H_   512
#define G_   2048      // 4*H
#define L_   2
#define NWG_ 16        // workgroups in recurrent phase; each owns 32 hidden units
#define REC_WLDS (128 * 512 * 2)          // 128KB bf16 weight chunk
#define REC_GLDS (4 * 32 * 32 * 4)        // 16KB gate-exchange buffer
#define REC_LDS  (REC_WLDS + REC_GLDS)

union Frag { v16bf v; u32x4 q[2]; };

__device__ __forceinline__ float sigm(float x) { return 1.0f / (1.0f + __expf(-x)); }

// ---------------------------------------------------------------------------
// Converters: x -> bf16; Wx -> WxT[N,K] bf16; Wh -> per-WG gate-major chunks;
// bias = bx + bh
// ---------------------------------------------------------------------------
__global__ void k_convert_x(const float* __restrict__ in, bf16* __restrict__ out, int n) {
    int i = blockIdx.x * blockDim.x + threadIdx.x;
    if (i < n) out[i] = (bf16)in[i];
}

__global__ void k_prep_weights(const float* __restrict__ Wx, const float* __restrict__ Wh,
                               const float* __restrict__ bx, const float* __restrict__ bh,
                               bf16* __restrict__ wxt, bf16* __restrict__ whc,
                               float* __restrict__ biasb) {
    int idx = blockIdx.x * blockDim.x + threadIdx.x;
    const int total = L_ * G_ * D_;
    if (idx >= total) return;
    int k = idx % D_;
    int n = (idx / D_) % G_;
    int l = idx / (D_ * G_);
    // WxT[l][n][k] = Wx[l][k][n]   (so B-fragments load contiguous K)
    wxt[idx] = (bf16)Wx[((size_t)l * D_ + k) * G_ + n];
    // WhC[l][wg][r][k]: r = gate*32 + jj  -> original column gate*H + wg*32 + jj
    int wg = n >> 7, r = n & 127;
    int col = (r >> 5) * H_ + wg * 32 + (r & 31);
    whc[idx] = (bf16)Wh[((size_t)l * H_ + k) * G_ + col];
    if (k == 0) biasb[l * G_ + n] = bx[l * G_ + n] + bh[l * G_ + n];
}

// ---------------------------------------------------------------------------
// xg = A[32768,512](bf16) @ WxT^T + bias  -> f32 [32768, 2048]
// Block 256 thr = 8 waves; block tile 128M x 128N; wave tile 64M x 32N.
// ---------------------------------------------------------------------------
__global__ __launch_bounds__(256) void k_gemm_xg(const bf16* __restrict__ A,
                                                 const bf16* __restrict__ Wt,
                                                 const float* __restrict__ biasb,
                                                 float* __restrict__ xg) {
    const int lane = threadIdx.x & 31, w = threadIdx.x >> 5;
    const int hi = lane >> 4, ln = lane & 15;
    const int mbase = blockIdx.x * 128 + (w & 1) * 64;
    const int nbase = blockIdx.y * 128 + (w >> 1) * 32;

    v8f acc[4][2];
#pragma unroll
    for (int nt = 0; nt < 2; ++nt) {
        float bv = biasb[nbase + nt * 16 + ln];
#pragma unroll
        for (int mt = 0; mt < 4; ++mt)
#pragma unroll
            for (int vv = 0; vv < 8; ++vv) acc[mt][nt][vv] = bv;
    }

    for (int k0 = 0; k0 < D_; k0 += 32) {
        Frag af[4], bfr[2];
#pragma unroll
        for (int mt = 0; mt < 4; ++mt) {
            const bf16* p = A + (size_t)(mbase + mt * 16 + ln) * D_ + k0 + hi * 8;
            af[mt].q[0] = *(const u32x4*)p;
            af[mt].q[1] = *(const u32x4*)(p + 16);
        }
#pragma unroll
        for (int nt = 0; nt < 2; ++nt) {
            const bf16* p = Wt + (size_t)(nbase + nt * 16 + ln) * D_ + k0 + hi * 8;
            bfr[nt].q[0] = *(const u32x4*)p;
            bfr[nt].q[1] = *(const u32x4*)(p + 16);
        }
#pragma unroll
        for (int mt = 0; mt < 4; ++mt)
#pragma unroll
            for (int nt = 0; nt < 2; ++nt)
                acc[mt][nt] = __builtin_amdgcn_wmma_f32_16x16x32_bf16(
                    false, af[mt].v, false, bfr[nt].v, (short)0, acc[mt][nt], false, false);
    }

#pragma unroll
    for (int mt = 0; mt < 4; ++mt)
#pragma unroll
        for (int nt = 0; nt < 2; ++nt)
#pragma unroll
            for (int vv = 0; vv < 8; ++vv) {
                int row = mbase + mt * 16 + vv + hi * 8;
                int col = nbase + nt * 16 + ln;
                xg[(size_t)row * G_ + col] = acc[mt][nt][vv];
            }
}

// ---------------------------------------------------------------------------
// Persistent recurrent kernel. 16 WGs x 256 thr. WG wg owns hidden units
// [wg*32, wg*32+32): Wh chunk (4 gates x 32 cols x 512 K, bf16, [N,K] layout)
// TDM-loaded into LDS once. Per step: gates = xg_t + h @ Wh via WMMA, gate
// exchange through LDS, elementwise cell, device-scope barrier.
// ---------------------------------------------------------------------------
__global__ __launch_bounds__(256) void k_lstm_rec(
    const float* __restrict__ xg,      // [B*T, 2048] (bias pre-added)
    const bf16*  __restrict__ whcL,    // layer's chunks [NWG][128][512]
    bf16*        __restrict__ hbuf,    // [2][32][512] double-buffered h (bf16)
    unsigned*    __restrict__ bar,     // grid barrier counter (zeroed on host)
    float*       __restrict__ ys,      // [B,T,512] f32 out or nullptr
    bf16*        __restrict__ seqb,    // [B,T,512] bf16 out or nullptr
    float*       __restrict__ hout,    // d_out h section [B,L,H]
    float*       __restrict__ cout,    // d_out c section [B,L,H]
    int layer) {
    extern __shared__ char smem[];
    bf16*  wlds = (bf16*)smem;
    float* glds = (float*)(smem + REC_WLDS);

    const int tid  = threadIdx.x;
    const int lane = tid & 31, w = tid >> 5;
    const int hi = lane >> 4, ln = lane & 15;
    const int wg = blockIdx.x;
    const int j0 = wg * 32;
    const int gate = w >> 1;                   // wave's gate: 0=f 1=g 2=i 3=o
    const int jcol = (w & 1) * 16 + ln;        // 0..31 within gate
    const int ncol = gate * H_ + j0 + jcol;    // column in xg

    // --- stage weight chunk into LDS via Tensor Data Mover (wave 0 only) ---
    const bf16* chunk = whcL + (size_t)wg * 128 * 512;
    if (w == 0) {
        unsigned long long ga = (unsigned long long)(const void*)chunk;
        u32x4 g0 = {1u,                                   // count=1 (valid)
                    0u,                                   // lds_addr = 0
                    (unsigned)ga,
                    (unsigned)((ga >> 32) & 0x1FFFFFFull) | 0x80000000u}; // type=2
        i32x8 g1 = {(int)(1u << 16),        // data_size = 2B
                    (int)(512u << 16),      // tensor_dim0 = 512 (lo16 @ [31:16])
                    (int)(128u << 16),      // dim0 hi=0 | tensor_dim1 = 128
                    (int)(512u << 16),      // dim1 hi=0 | tile_dim0 = 512
                    128,                    // tile_dim1 = 128, tile_dim2 = 0
                    512,                    // tensor_dim0_stride = 512
                    0, 0};
        i32x4 gz4 = {0, 0, 0, 0};
        i32x8 gz8 = {0, 0, 0, 0, 0, 0, 0, 0};
        __builtin_amdgcn_tensor_load_to_lds(g0, g1, gz4, gz4, gz8, 0);
        __builtin_amdgcn_s_wait_tensorcnt(0);
    }
    __syncthreads();

    float creg[4] = {0.f, 0.f, 0.f, 0.f};     // cell state: 4 (b,jj) per thread

    for (int t = 0; t < T_; ++t) {
        const bf16* hsrc = hbuf + (size_t)(t & 1) * (B_ * H_);
        bf16*       hdst = hbuf + (size_t)((t + 1) & 1) * (B_ * H_);

        // accumulators seeded with xg_t (bias folded in)
        v8f acc[2];
#pragma unroll
        for (int mt = 0; mt < 2; ++mt)
#pragma unroll
            for (int vv = 0; vv < 8; ++vv) {
                int b = mt * 16 + vv + hi * 8;
                acc[mt][vv] = xg[((size_t)b * T_ + t) * G_ + ncol];
            }

        // h[32,512] @ WhChunk[:,wave cols] : 16 K-steps x 2 M-tiles
#pragma unroll 4
        for (int kk = 0; kk < 16; ++kk) {
            const int k0 = kk * 32;
            Frag bfr;
            const bf16* wp = wlds + ((w * 16 + ln) * 512 + k0 + hi * 8);
            bfr.q[0] = *(const u32x4*)wp;
            bfr.q[1] = *(const u32x4*)(wp + 16);
#pragma unroll
            for (int mt = 0; mt < 2; ++mt) {
                Frag af;
                const bf16* ap = hsrc + (size_t)(mt * 16 + ln) * H_ + k0 + hi * 8;
                af.q[0] = *(const u32x4*)ap;
                af.q[1] = *(const u32x4*)(ap + 16);
                acc[mt] = __builtin_amdgcn_wmma_f32_16x16x32_bf16(
                    false, af.v, false, bfr.v, (short)0, acc[mt], false, false);
            }
        }

        // publish gate preactivations to LDS for cross-wave cell update
#pragma unroll
        for (int mt = 0; mt < 2; ++mt)
#pragma unroll
            for (int vv = 0; vv < 8; ++vv) {
                int b = mt * 16 + vv + hi * 8;
                glds[(gate * 32 + b) * 32 + jcol] = acc[mt][vv];
            }
        __syncthreads();

        // elementwise LSTM cell: each thread owns 4 (b, jj) pairs
#pragma unroll
        for (int q = 0; q < 4; ++q) {
            int e = q * 256 + tid;
            int b = e >> 5, jj = e & 31;
            float f = sigm(glds[(0 * 32 + b) * 32 + jj]);
            float g = tanhf(glds[(1 * 32 + b) * 32 + jj]);
            float i = sigm(glds[(2 * 32 + b) * 32 + jj]);
            float o = sigm(glds[(3 * 32 + b) * 32 + jj]);
            float c = f * creg[q] + i * g;
            creg[q] = c;
            float h = o * tanhf(c);
            hdst[b * H_ + j0 + jj] = (bf16)h;
            if (ys)   ys[((size_t)b * T_ + t) * H_ + j0 + jj] = h;
            if (seqb) seqb[((size_t)b * T_ + t) * H_ + j0 + jj] = (bf16)h;
            if (t == T_ - 1) {
                hout[(b * L_ + layer) * H_ + j0 + jj] = h;
                cout[(b * L_ + layer) * H_ + j0 + jj] = c;
            }
        }

        // device-scope grid barrier (16 resident WGs)
        __threadfence();
        __syncthreads();
        if (tid == 0) {
            __hip_atomic_fetch_add(bar, 1u, __ATOMIC_RELEASE, __HIP_MEMORY_SCOPE_AGENT);
            unsigned tgt = (unsigned)(t + 1) * gridDim.x;
            while (__hip_atomic_load(bar, __ATOMIC_ACQUIRE, __HIP_MEMORY_SCOPE_AGENT) < tgt)
                __builtin_amdgcn_s_sleep(2);
        }
        __syncthreads();
        __threadfence();
    }
}

// ---------------------------------------------------------------------------
// Host side
// ---------------------------------------------------------------------------
extern "C" void kernel_launch(void* const* d_in, const int* in_sizes, int n_in,
                              void* d_out, int out_size, void* d_ws, size_t ws_size,
                              hipStream_t stream) {
    const float* x  = (const float*)d_in[0];
    const float* Wx = (const float*)d_in[1];
    const float* bx = (const float*)d_in[2];
    const float* Wh = (const float*)d_in[3];
    const float* bh = (const float*)d_in[4];

    char* ws = (char*)d_ws;
    const size_t off_xg    = 0;                                       // f32 [B*T,2048]
    const size_t off_xbf   = off_xg    + (size_t)B_ * T_ * G_ * 4;    // bf16 [B*T,512]
    const size_t off_seqbf = off_xbf   + (size_t)B_ * T_ * D_ * 2;    // bf16 [B*T,512]
    const size_t off_wxt   = off_seqbf + (size_t)B_ * T_ * H_ * 2;    // bf16 [L,2048,512]
    const size_t off_whc   = off_wxt   + (size_t)L_ * G_ * D_ * 2;    // bf16 [L,16,128,512]
    const size_t off_bias  = off_whc   + (size_t)L_ * G_ * D_ * 2;    // f32 [L,2048]
    const size_t off_hbuf  = off_bias  + (size_t)L_ * G_ * 4;         // bf16 [2,32,512]
    const size_t off_bar   = off_hbuf  + (size_t)2 * B_ * H_ * 2;

    float*    xg    = (float*)(ws + off_xg);
    bf16*     xbf   = (bf16*)(ws + off_xbf);
    bf16*     seqbf = (bf16*)(ws + off_seqbf);
    bf16*     wxt   = (bf16*)(ws + off_wxt);
    bf16*     whc   = (bf16*)(ws + off_whc);
    float*    biasb = (float*)(ws + off_bias);
    bf16*     hbuf  = (bf16*)(ws + off_hbuf);
    unsigned* bar   = (unsigned*)(ws + off_bar);

    float* seq_out = (float*)d_out;                         // [B,T,H]
    float* hout    = seq_out + (size_t)B_ * T_ * H_;        // [B,L,H]
    float* cout    = hout    + (size_t)B_ * L_ * H_;        // [B,L,H]

    // precision conversion + weight transposition/packing
    {
        int n = B_ * T_ * D_;
        k_convert_x<<<(n + 255) / 256, 256, 0, stream>>>(x, xbf, n);
        int nw = L_ * G_ * D_;
        k_prep_weights<<<(nw + 255) / 256, 256, 0, stream>>>(Wx, Wh, bx, bh, wxt, whc, biasb);
    }

    for (int l = 0; l < L_; ++l) {
        const bf16* A = (l == 0) ? xbf : seqbf;
        k_gemm_xg<<<dim3((B_ * T_) / 128, G_ / 128), 256, 0, stream>>>(
            A, wxt + (size_t)l * G_ * D_, biasb + (size_t)l * G_, xg);
        (void)hipMemsetAsync(bar, 0, 64, stream);
        (void)hipMemsetAsync(hbuf, 0, (size_t)2 * B_ * H_ * sizeof(bf16), stream);
        k_lstm_rec<<<NWG_, 256, REC_LDS, stream>>>(
            xg, whc + (size_t)l * NWG_ * 128 * 512, hbuf, bar,
            (l == L_ - 1) ? seq_out : nullptr,
            (l == 0) ? seqbf : nullptr,
            hout, cout, l);
    }
}
